// GraphConvEncoder_20332375179522
// MI455X (gfx1250) — matrix operand order repaired and checked by reading, not compiled, not run on hardware
//
#include <hip/hip_runtime.h>
#include <stdint.h>
#include <stddef.h>

// Problem constants (match reference)
#define NN 256      // boxes per item (image H=W)
#define MM 256      // feature dim / output channels
#define NP 258      // padded (halo=1) image dimension
#define BATCH 2
#define NBLOCKS 2

typedef __attribute__((ext_vector_type(16))) __bf16 v16bf;
typedef __attribute__((ext_vector_type(8)))  float  v8f;

struct alignas(16) U4 { unsigned int x, y, z, w; };

static __device__ __forceinline__ unsigned short f2bf(float f) {
  union { float f; unsigned int u; } v; v.f = f;
  unsigned int u = v.u;
  u += 0x7fffu + ((u >> 16) & 1u);          // round-to-nearest-even
  return (unsigned short)(u >> 16);
}
static __device__ __forceinline__ float bf2f(unsigned short h) {
  union { unsigned int u; float f; } v; v.u = ((unsigned int)h) << 16;
  return v.f;
}

// ---------------------------------------------------------------- zero scratch
__global__ void zero_u4(U4* __restrict__ p, long long n) {
  long long i = (long long)blockIdx.x * blockDim.x + threadIdx.x;
  long long stride = (long long)gridDim.x * blockDim.x;
  U4 z; z.x = z.y = z.z = z.w = 0u;
  for (; i < n; i += stride) p[i] = z;
}

// ------------------------------------------------- pack weights into B-frags
// Source: w[O=256][CIN][3][3] f32 (OIHW).
// Dest  : per fragment (tap, ck, kt): 32 lanes x 16 bf16 (32B) laid out in the
//         v_wmma_16x16x32_bf16 B layout: lane = 16*((k/8)%2) + n, elem s = 8*(k/16) + k%8.
template<int CIN>
__global__ void pack_w(const float* __restrict__ w, U4* __restrict__ wp) {
  constexpr int CHUNKS = CIN / 32;
  int id = blockIdx.x * blockDim.x + threadIdx.x;
  if (id >= 9 * CHUNKS * 16 * 32) return;
  int lane = id & 31;  int f = id >> 5;       // f = (tap*CHUNKS+ck)*16 + kt
  int kt = f & 15;     int fc = f >> 4;
  int ck = fc % CHUNKS; int tap = fc / CHUNKS;
  int n = lane & 15, grp = lane >> 4;
  int kout = kt * 16 + n, ty = tap / 3, tx = tap % 3;
  union { unsigned short s[16]; U4 q[2]; } o;
  #pragma unroll
  for (int v = 0; v < 8; ++v)
    #pragma unroll
    for (int h = 0; h < 2; ++h) {
      int c = ck * 32 + grp * 8 + ((2 * v + h) & 7) + (((2 * v + h) >> 3) << 4);
      o.s[2 * v + h] = f2bf(w[((size_t)(kout * CIN + c) * 3 + ty) * 3 + tx]);
    }
  U4* dst = wp + (size_t)id * 2;
  dst[0] = o.q[0]; dst[1] = o.q[1];
}

// ------------------------------------------------------------- adapter (1x1, C=7)
// edge[m,i,j] = sum_c e[i,j,c]*aw[m,c] + ab[m]  -> bf16 [p][M]
__global__ void adapter_k(const float* __restrict__ e, const float* __restrict__ aw,
                          const float* __restrict__ ab, unsigned short* __restrict__ out) {
  int id = blockIdx.x * blockDim.x + threadIdx.x;
  if (id >= NN * NN * 32) return;
  int p = id >> 5, q = id & 31;
  int m0 = q * 8;
  float ev[7];
  const float* ep = e + (size_t)p * 7;
  #pragma unroll
  for (int c = 0; c < 7; ++c) ev[c] = ep[c];
  union { unsigned short s[8]; U4 q4; } o;
  #pragma unroll
  for (int mi = 0; mi < 8; ++mi) {
    int m = m0 + mi;
    float s = ab[m];
    #pragma unroll
    for (int c = 0; c < 7; ++c) s += ev[c] * aw[m * 7 + c];
    o.s[mi] = f2bf(s);
  }
  *(U4*)(out + (size_t)p * MM + m0) = o.q4;
}

// ---------------------------------------------- build concat [node_i|edge|node_j]
// Writes padded bf16 [NP][NP][768] interior; halo stays zero.
__global__ void concat_k(const float* __restrict__ node, const unsigned short* __restrict__ edge,
                         unsigned short* __restrict__ cp) {
  int id = blockIdx.x * blockDim.x + threadIdx.x;
  if (id >= NN * NN * 96) return;
  int p = id / 96, q = id % 96;
  int c0 = q * 8;
  int i = p >> 8, j = p & 255;
  unsigned short* dst = cp + ((size_t)(i + 1) * NP + (j + 1)) * 768 + c0;
  union { unsigned short s[8]; U4 q4; } o;
  if (c0 >= 256 && c0 < 512) {
    o.q4 = *(const U4*)(edge + (size_t)p * MM + (c0 - 256));
  } else {
    const float* src = (c0 < 256) ? (node + (size_t)i * MM + c0)
                                  : (node + (size_t)j * MM + (c0 - 512));
    #pragma unroll
    for (int t = 0; t < 8; ++t) o.s[t] = f2bf(src[t]);
  }
  *(U4*)dst = o.q4;
}

// ------------------------------------------------------- 3x3 conv via WMMA bf16
// One wave computes 32 pixels x 64 output channels (2 pixel-subtiles x 4 kt
// tiles): each B fragment feeds 2 WMMAs, each A fragment feeds 4 WMMAs
// -> 0.75 KB loaded per 16 KFLOP WMMA (~21 FLOP/byte from L2).
// Input: padded bf16 [NP][NP][CIN]; weights pre-packed B-frags; bias f32.
// Sinks (nullable): padded bf16 interior, flat bf16 [p][M], flat f32 [p][M].
template<int CIN>
__global__ __launch_bounds__(256) void conv3x3_k(
    const unsigned short* __restrict__ inPad,
    const U4* __restrict__ wp,
    const float* __restrict__ bias,
    unsigned short* __restrict__ outPad,
    unsigned short* __restrict__ outFlatB,
    float* __restrict__ outFlatF) {
  constexpr int CHUNKS = CIN / 32;
  int tid  = blockIdx.x * blockDim.x + threadIdx.x;
  int wid  = tid >> 5;
  int lane = tid & 31;
  int ktg  = wid & 3;            // which group of 4 kout-tiles (64 channels)
  int pt   = wid >> 2;           // pixel tile-pair: 32 pixels of one row
  int r    = pt >> 3;
  int col0 = (pt & 7) << 5;
  int ln15 = lane & 15;
  int grp  = lane >> 4;

  v8f acc0[4], acc1[4];
  {
    v8f z = {};
    #pragma unroll
    for (int t = 0; t < 4; ++t) { acc0[t] = z; acc1[t] = z; }
  }

  for (int tap = 0; tap < 9; ++tap) {
    int dy = tap / 3 - 1, dx = tap % 3 - 1;
    const unsigned short* arow0 =
        inPad + ((size_t)(r + dy + 1) * NP + (size_t)(col0 + dx + 1 + ln15)) * CIN + grp * 8;
    const unsigned short* arow1 = arow0 + (size_t)16 * CIN;
    if (tap < 8) {  // prefetch next tap's row (global_prefetch_b8)
      int dy2 = (tap + 1) / 3 - 1, dx2 = (tap + 1) % 3 - 1;
      __builtin_prefetch(
          inPad + ((size_t)(r + dy2 + 1) * NP + (size_t)(col0 + dx2 + 1 + ln15)) * CIN + grp * 8,
          0, 1);
    }
    const U4* wtap = wp + ((size_t)tap * CHUNKS * 16 * 64) + (size_t)lane * 2;
    for (int ck = 0; ck < CHUNKS; ++ck) {
      union { U4 u[2]; v16bf v; } a0, a1;
      a0.u[0] = *(const U4*)(arow0 + ck * 32);        // K = c0+8g .. +7
      a0.u[1] = *(const U4*)(arow0 + ck * 32 + 16);   // K = c0+16+8g .. +7
      a1.u[0] = *(const U4*)(arow1 + ck * 32);
      a1.u[1] = *(const U4*)(arow1 + ck * 32 + 16);
      const U4* wck = wtap + ((size_t)ck * 16 + ktg * 4) * 64;
      #pragma unroll
      for (int t = 0; t < 4; ++t) {
        union { U4 u[2]; v16bf v; } b;
        b.u[0] = wck[t * 64];
        b.u[1] = wck[t * 64 + 1];
        acc0[t] = __builtin_amdgcn_wmma_f32_16x16x32_bf16(
            false, a0.v, false, b.v, (short)0, acc0[t], false, false);
        acc1[t] = __builtin_amdgcn_wmma_f32_16x16x32_bf16(
            false, a1.v, false, b.v, (short)0, acc1[t], false, false);
      }
    }
  }

  const bool doPad = outPad   != nullptr;
  const bool doFB  = outFlatB != nullptr;
  const bool doFF  = outFlatF != nullptr;
  #pragma unroll
  for (int t = 0; t < 4; ++t) {
    int kout = (ktg * 4 + t) * 16 + ln15;       // C/D layout: n = lane%16
    float bv = bias[kout];
    #pragma unroll
    for (int v = 0; v < 8; ++v) {
      int mrow = v + grp * 8;                   // C/D layout: m = vgpr + 8*(lane/16)
      #pragma unroll
      for (int half = 0; half < 2; ++half) {
        float val = (half ? acc1[t][v] : acc0[t][v]) + bv;
        val = fmaxf(val, 0.0f);                 // relu
        int col = col0 + half * 16 + mrow;
        if (doPad) outPad[((size_t)(r + 1) * NP + (col + 1)) * MM + kout] = f2bf(val);
        if (doFB)  outFlatB[((size_t)r * NN + col) * MM + kout] = f2bf(val);
        if (doFF)  outFlatF[((size_t)r * NN + col) * MM + kout] = val;
      }
    }
  }
}

// -------------------------------- attention: 1x1 conv -> leaky -> softmax -> agg
// One 256-thread workgroup per row i. feat is padded bf16 [NP][NP][M].
__global__ __launch_bounds__(256) void att_k(const unsigned short* __restrict__ featPad,
                                             const float* __restrict__ attw,
                                             const float* __restrict__ attb,
                                             float* __restrict__ nodeOut) {
  __shared__ float red[256];
  __shared__ float coeff[256];
  int i = blockIdx.x;
  int j = threadIdx.x;
  const unsigned short* fp = featPad + ((size_t)(i + 1) * NP + (j + 1)) * MM;
  float a = attb[0];
  for (int m = 0; m < MM; ++m) a += bf2f(fp[m]) * attw[m];
  a = a > 0.0f ? a : 0.01f * a;                 // leaky_relu(0.01)
  // softmax over j
  red[j] = a; __syncthreads();
  for (int s = 128; s > 0; s >>= 1) { if (j < s) red[j] = fmaxf(red[j], red[j + s]); __syncthreads(); }
  float mx = red[0]; __syncthreads();
  float e = __expf(a - mx);
  red[j] = e; __syncthreads();
  for (int s = 128; s > 0; s >>= 1) { if (j < s) red[j] += red[j + s]; __syncthreads(); }
  float sum = red[0];
  coeff[j] = e / sum;
  __syncthreads();
  // node[i][m] = sum_j coeff[j] * feat[(i,j)][m]; thread handles m = j
  int m = j;
  float accv = 0.0f;
  const unsigned short* base = featPad + ((size_t)(i + 1) * NP + 1) * MM + m;
  for (int jj = 0; jj < NN; ++jj) accv += coeff[jj] * bf2f(base[(size_t)jj * MM]);
  nodeOut[(size_t)i * MM + m] = accv;
}

// ---------------------------------------------------------------------- launch
extern "C" void kernel_launch(void* const* d_in, const int* in_sizes, int n_in,
                              void* d_out, int out_size, void* d_ws, size_t ws_size,
                              hipStream_t stream) {
  (void)in_sizes; (void)n_in; (void)out_size;
  const float* edge_feats = (const float*)d_in[0];   // [B,N,N,7]
  const float* node_feats = (const float*)d_in[1];   // [B,N,M]
  const float* adapter_w  = (const float*)d_in[2];   // [M,7]
  const float* adapter_b  = (const float*)d_in[3];   // [M]
  const float* conv0_w    = (const float*)d_in[4];   // [M,3M,3,3]
  const float* conv0_b    = (const float*)d_in[5];
  const float* conv1_w    = (const float*)d_in[6];   // [M,M,3,3]
  const float* conv1_b    = (const float*)d_in[7];
  const float* last_w     = (const float*)d_in[8];   // [M,M,3,3]
  const float* last_b     = (const float*)d_in[9];
  const float* att_w      = (const float*)d_in[10];  // [M]
  const float* att_b      = (const float*)d_in[11];  // [1]

  float* out_node = (float*)d_out;                          // [B][N][M]
  float* out_edge = out_node + (size_t)BATCH * NN * MM;     // [B][N][N][M]

  // Scratch layout (all sizes multiples of 256B)
  const size_t szConcat = (size_t)NP * NP * 768 * 2;        // 102,242,304
  const size_t szPadM   = (size_t)NP * NP * MM  * 2;        //  34,080,768
  const size_t szEdge   = (size_t)NN * NN * MM  * 2;        //  33,554,432
  const size_t szNode   = (size_t)NN * MM * 4;              //     262,144
  const size_t szW0     = (size_t)9 * 24 * 16 * 1024;       //   3,538,944
  const size_t szW1     = (size_t)9 * 8  * 16 * 1024;       //   1,179,648
  const size_t need = szConcat + 2 * szPadM + szEdge + szNode + szW0 + 2 * szW1;
  if (ws_size < need) return;

  char* ws = (char*)d_ws;
  unsigned short* concatPad = (unsigned short*)(ws);
  unsigned short* y0Pad     = (unsigned short*)(ws + szConcat);
  unsigned short* featPad   = (unsigned short*)(ws + szConcat + szPadM);
  unsigned short* edgeB     = (unsigned short*)(ws + szConcat + 2 * szPadM);
  float*          nodeB     = (float*)(ws + szConcat + 2 * szPadM + szEdge);
  U4*             w0p       = (U4*)(ws + szConcat + 2 * szPadM + szEdge + szNode);
  U4*             w1p       = (U4*)((char*)w0p + szW0);
  U4*             wlp       = (U4*)((char*)w1p + szW1);

  // Zero the three padded buffers (halos must be 0 for SAME padding)
  long long zeroN = (long long)(szConcat + 2 * szPadM) / 16;
  zero_u4<<<2048, 256, 0, stream>>>((U4*)ws, zeroN);

  // Repack weights into WMMA B-fragment layout (bf16)
  pack_w<768><<<(9 * 24 * 16 * 32 + 255) / 256, 256, 0, stream>>>(conv0_w, w0p);
  pack_w<256><<<(9 * 8  * 16 * 32 + 255) / 256, 256, 0, stream>>>(conv1_w, w1p);
  pack_w<256><<<(9 * 8  * 16 * 32 + 255) / 256, 256, 0, stream>>>(last_w,  wlp);

  // 2048 pixel-tile-pairs (32 px each) x 4 ktg waves = 8192 waves = 1024 WGs
  const int convBlocks = (NN * NN / 32) * 4 * 32 / 256;
  for (int b = 0; b < BATCH; ++b) {
    adapter_k<<<NN * NN * 32 / 256, 256, 0, stream>>>(
        edge_feats + (size_t)b * NN * NN * 7, adapter_w, adapter_b, edgeB);
    const float* nodeCur = node_feats + (size_t)b * NN * MM;
    for (int blk = 0; blk < NBLOCKS; ++blk) {
      const bool lastBlk = (blk == NBLOCKS - 1);
      concat_k<<<NN * NN * 96 / 256, 256, 0, stream>>>(nodeCur, edgeB, concatPad);
      conv3x3_k<768><<<convBlocks, 256, 0, stream>>>(
          concatPad, w0p, conv0_b, y0Pad, (unsigned short*)nullptr, (float*)nullptr);
      conv3x3_k<256><<<convBlocks, 256, 0, stream>>>(
          y0Pad, w1p, conv1_b, featPad, (unsigned short*)nullptr, (float*)nullptr);
      float* nodeDst = lastBlk ? (out_node + (size_t)b * NN * MM) : nodeB;
      att_k<<<NN, 256, 0, stream>>>(featPad, att_w, att_b, nodeDst);
      if (lastBlk)
        conv3x3_k<256><<<convBlocks, 256, 0, stream>>>(
            featPad, wlp, last_b, (unsigned short*)nullptr, (unsigned short*)nullptr,
            out_edge + (size_t)b * NN * NN * MM);
      else
        conv3x3_k<256><<<convBlocks, 256, 0, stream>>>(
            featPad, wlp, last_b, (unsigned short*)nullptr, edgeB, (float*)nullptr);
      nodeCur = nodeDst;
    }
  }
}
